// CiaoSR_386547057372
// MI455X (gfx1250) — matrix operand order repaired and checked by reading, not compiled, not run on hardware
//
#include <hip/hip_runtime.h>
#include <math.h>

typedef __attribute__((ext_vector_type(16))) _Float16 v16h;
typedef __attribute__((ext_vector_type(8)))  float    v8f;
typedef __attribute__((ext_vector_type(4)))  unsigned u32x4;

union F16x16 { v16h v; unsigned u[8]; _Float16 h[16]; };

__device__ __forceinline__ v8f wmma_f16(v16h a, v16h b, v8f c) {
  return __builtin_amdgcn_wmma_f32_16x16x32_f16(
      /*neg_a=*/false, a, /*neg_b=*/false, b,
      /*c_mod=*/(short)0, c, /*reuse_a=*/false, /*reuse_b=*/false);
}

// A fragment from a pair-packed row (u32 = 2 f16 along K), row32 already at
// this ktile's 16 pairs. Two contiguous 16B runs -> two 128-bit loads.
__device__ __forceinline__ v16h loadA_row(const unsigned* row32, int hb) {
  u32x4 a = *(const u32x4*)(row32 + hb * 8);
  u32x4 b = *(const u32x4*)(row32 + hb * 8 + 4);
  F16x16 A;
  A.u[0] = a[0]; A.u[1] = a[1]; A.u[2] = a[2]; A.u[3] = a[3];
  A.u[4] = b[0]; A.u[5] = b[1]; A.u[6] = b[2]; A.u[7] = b[3];
  return A.v;
}

// B fragment from a fragment-swizzled weight buffer [ktile][ntile][lane][8u32]
__device__ __forceinline__ v16h loadB_swz(const unsigned* frags, int NT,
                                          int kt, int nt, int lane) {
  const unsigned* p = frags + (((kt * NT + nt) * 32) + lane) * 8;
  u32x4 a = *(const u32x4*)(p);
  u32x4 b = *(const u32x4*)(p + 4);
  F16x16 B;
  B.u[0] = a[0]; B.u[1] = a[1]; B.u[2] = a[2]; B.u[3] = a[3];
  B.u[4] = b[0]; B.u[5] = b[1]; B.u[6] = b[2]; B.u[7] = b[3];
  return B.v;
}

__device__ __forceinline__ unsigned pk2(float a, float b) {
  _Float16 ha = (_Float16)a, hb = (_Float16)b;
  unsigned short ua = __builtin_bit_cast(unsigned short, ha);
  unsigned short ub = __builtin_bit_cast(unsigned short, hb);
  return (unsigned)ua | ((unsigned)ub << 16);
}

// ---------------------------------------------------------------------------
// Kernel 0: weight prep -> f16, pre-swizzled into B-fragment register order.
// Fragment element (kt,nt,lane,i) holds pair (W[k][n], W[k+1][n]) with
//   k = 2*(kt*16 + (lane>>4)*8 + i), n = nt*16 + (lane&15)
// ---------------------------------------------------------------------------
__global__ void prep_weights(const float* __restrict__ coef_w,
                             const float* __restrict__ freq_w,
                             const float* __restrict__ phase_w,
                             const float* __restrict__ im_w1,
                             const float* __restrict__ im_w2,
                             const float* __restrict__ kk_w1,
                             const float* __restrict__ kk_w2,
                             const float* __restrict__ vv_w1,
                             const float* __restrict__ vv_w2,
                             unsigned* __restrict__ cwp,
                             unsigned* __restrict__ w1p,
                             unsigned* __restrict__ w2p,
                             unsigned* __restrict__ w1qp,
                             unsigned* __restrict__ w2qp) {
  int e = blockIdx.x * blockDim.x + threadIdx.x;
  const int N_CW = 18 * 10 * 256, N_W1 = 2 * 16 * 256, N_W2 = 8 * 4 * 256;
  const int N_W1Q = 3 * 16 * 256, N_W2Q = 8 * 36 * 256;

  if (e < N_CW) {
    int frag = e >> 8, rem = e & 255, lane = rem >> 3, i = rem & 7;
    int kt = frag / 10, nt = frag % 10;
    int k = 2 * (kt * 16 + (lane >> 4) * 8 + i);  // even, pair in same tap
    int n = nt * 16 + (lane & 15);
    int tap = k >> 6, cin = k & 63;
    float a, b;
    if (n < 64) {
      a = coef_w[(n * 64 + cin) * 9 + tap];
      b = coef_w[(n * 64 + cin + 1) * 9 + tap];
    } else if (n < 128) {
      int o = n - 64;
      a = freq_w[(o * 64 + cin) * 9 + tap];
      b = freq_w[(o * 64 + cin + 1) * 9 + tap];
    } else {
      int o = n - 128;
      a = phase_w[(o * 64 + cin) * 9 + tap];
      b = phase_w[(o * 64 + cin + 1) * 9 + tap];
    }
    cwp[e] = pk2(a, b);
    return;
  }
  e -= N_CW;
  if (e < N_W1) {
    int frag = e >> 8, rem = e & 255, lane = rem >> 3, i = rem & 7;
    int kt = frag / 16, nt = frag % 16;
    int k = 2 * (kt * 16 + (lane >> 4) * 8 + i);
    int n = nt * 16 + (lane & 15);
    w1p[e] = pk2(im_w1[k * 256 + n], im_w1[(k + 1) * 256 + n]);
    return;
  }
  e -= N_W1;
  if (e < N_W2) {
    int frag = e >> 8, rem = e & 255, lane = rem >> 3, i = rem & 7;
    int kt = frag / 4, nt = frag % 4;
    int k = 2 * (kt * 16 + (lane >> 4) * 8 + i);
    int n = nt * 16 + (lane & 15);
    w2p[e] = pk2(im_w2[k * 64 + n], im_w2[(k + 1) * 64 + n]);
    return;
  }
  e -= N_W2;
  if (e < 2 * N_W1Q) {
    int g = e / N_W1Q, r = e % N_W1Q;
    int frag = r >> 8, rem = r & 255, lane = rem >> 3, i = rem & 7;
    int kt = frag / 16, nt = frag % 16;
    int k = 2 * (kt * 16 + (lane >> 4) * 8 + i);
    int n = nt * 16 + (lane & 15);
    const float* w = g ? vv_w1 : kk_w1;
    float a = (k < 68) ? w[k * 256 + n] : 0.f;
    float b = (k + 1 < 68) ? w[(k + 1) * 256 + n] : 0.f;
    w1qp[e] = pk2(a, b);
    return;
  }
  e -= 2 * N_W1Q;
  if (e < 2 * N_W2Q) {
    int g = e / N_W2Q, r = e % N_W2Q;
    int frag = r >> 8, rem = r & 255, lane = rem >> 3, i = rem & 7;
    int kt = frag / 36, nt = frag % 36;
    int k = 2 * (kt * 16 + (lane >> 4) * 8 + i);
    int n = nt * 16 + (lane & 15);
    const float* w = g ? vv_w2 : kk_w2;
    w2qp[e] = pk2(w[k * 576 + n], w[(k + 1) * 576 + n]);
    return;
  }
}

// ---------------------------------------------------------------------------
// Kernel 1: im2col conv GEMM (M=16 pixels, K=576, N=160) via WMMA + LTE.
// ---------------------------------------------------------------------------
__global__ __launch_bounds__(32, 1) void conv_lte(
    const float* __restrict__ feat, const unsigned* __restrict__ cwp,
    const float* __restrict__ coef_b, const float* __restrict__ freq_b,
    const float* __restrict__ phase_b, _Float16* __restrict__ x16) {
  __shared__ float sacc[16 * 160];
  const int lane = threadIdx.x;
  const int m = lane & 15, hb = lane >> 4;
  const int t = blockIdx.x;
  const int y = t >> 3;
  const int xb = (t & 7) << 4;
  const int px = xb + m;

  v8f acc[10];
#pragma unroll
  for (int nt = 0; nt < 10; ++nt)
    acc[nt] = (v8f){0.f, 0.f, 0.f, 0.f, 0.f, 0.f, 0.f, 0.f};

  for (int kt = 0; kt < 18; ++kt) {  // K = 576 = 18 * 32
    F16x16 A;
    const int tap = kt >> 1;  // 32-K step spans half a 64-chan tap
    const int dy = tap / 3 - 1, dx = tap % 3 - 1;
    const int yy = y + dy, xx = px + dx;
    const bool in = (yy >= 0) & (yy < 128) & (xx >= 0) & (xx < 128);
    const float* src = feat + yy * 128 + xx;
#pragma unroll
    for (int i = 0; i < 8; ++i) {
      int Kb = hb * 16 + 2 * i;
      int cin = ((kt & 1) << 5) + Kb;  // (kt*32+Kb) & 63
      float f0 = in ? src[cin * 16384] : 0.f;
      float f1 = in ? src[(cin + 1) * 16384] : 0.f;
      A.h[2 * i] = (_Float16)f0;
      A.h[2 * i + 1] = (_Float16)f1;
    }
    v16h Bf[10];
#pragma unroll
    for (int nt = 0; nt < 10; ++nt) Bf[nt] = loadB_swz(cwp, 10, kt, nt, lane);
#pragma unroll
    for (int nt = 0; nt < 10; ++nt) acc[nt] = wmma_f16(A.v, Bf[nt], acc[nt]);
  }
  // dump C/D tiles -> LDS [pixel][outchan] f32
#pragma unroll
  for (int nt = 0; nt < 10; ++nt)
#pragma unroll
    for (int r = 0; r < 8; ++r)
      sacc[(hb * 8 + r) * 160 + nt * 16 + m] = acc[nt][r];
  __syncthreads();

  // LTE post: lane handles pixel m, channel half hb (32 channels)
  const float* row = &sacc[m * 160];
  const int p = y * 128 + px;
  const float PI = 3.14159265358979323846f;
#pragma unroll 4
  for (int j = 0; j < 32; ++j) {
    float f0 = row[64 + 2 * j] + freq_b[2 * j];
    float f1 = row[64 + 2 * j + 1] + freq_b[2 * j + 1];
    float ph = row[128 + j] + phase_b[j];
    float qq = PI * (f0 + f1 + ph);
    int c = hb * 32 + j;
    float cf = row[c] + coef_b[c];
    float v = cf * (hb ? __sinf(qq) : __cosf(qq));
    x16[p * 64 + c] = (_Float16)v;
  }
}

// ---------------------------------------------------------------------------
// Kernel 2: imnet MLP 64 -> 256 (relu) -> 64 over 16384 rows. One wave per
// 16-row tile. feat16[p][c] stored f16 row-major for the gather stage.
// ---------------------------------------------------------------------------
__global__ __launch_bounds__(32) void imnet_mlp(
    const _Float16* __restrict__ x16, const unsigned* __restrict__ w1p,
    const unsigned* __restrict__ w2p, const float* __restrict__ b1,
    const float* __restrict__ b2, _Float16* __restrict__ feat16) {
  __shared__ unsigned hidu[16 * 128];  // 16 x 256 f16
  _Float16* hid = (_Float16*)hidu;
  const int lane = threadIdx.x;
  const int m = lane & 15, hb = lane >> 4;
  const int t = blockIdx.x;
  const unsigned* xr = (const unsigned*)x16;  // [p][32] u32

  v16h a1[2];
#pragma unroll
  for (int kt = 0; kt < 2; ++kt)
    a1[kt] = loadA_row(xr + (t * 16 + m) * 32 + kt * 16, hb);

  for (int nt = 0; nt < 16; ++nt) {
    v8f acc = (v8f){0.f, 0.f, 0.f, 0.f, 0.f, 0.f, 0.f, 0.f};
#pragma unroll
    for (int kt = 0; kt < 2; ++kt)
      acc = wmma_f16(a1[kt], loadB_swz(w1p, 16, kt, nt, lane), acc);
    float bb = b1[nt * 16 + m];
#pragma unroll
    for (int r = 0; r < 8; ++r)
      hid[(hb * 8 + r) * 256 + nt * 16 + m] = (_Float16)fmaxf(acc[r] + bb, 0.f);
  }
  __syncthreads();

  v16h a2[8];
#pragma unroll
  for (int kt = 0; kt < 8; ++kt)
    a2[kt] = loadA_row(hidu + m * 128 + kt * 16, hb);

  for (int nt = 0; nt < 4; ++nt) {
    v8f acc = (v8f){0.f, 0.f, 0.f, 0.f, 0.f, 0.f, 0.f, 0.f};
#pragma unroll
    for (int kt = 0; kt < 8; ++kt)
      acc = wmma_f16(a2[kt], loadB_swz(w2p, 4, kt, nt, lane), acc);
    float bb = b2[nt * 16 + m];
#pragma unroll
    for (int r = 0; r < 8; ++r)
      feat16[(t * 16 + hb * 8 + r) * 64 + nt * 16 + m] = (_Float16)(acc[r] + bb);
  }
}

// ---------------------------------------------------------------------------
// Kernel 3: query MLP. Grid (2048, 2), one wave per (32-query group, net).
// 2 M-tiles per wave -> every B fragment feeds 4 WMMAs (2 samples x 2 tiles),
// halving the dominant W2 L2 stream. B double-buffered in the k-loop.
// __launch_bounds__(32,1): license the full register file so the 32 A2
// fragments (256 VGPRs) stay register-resident with no scratch spills.
// ---------------------------------------------------------------------------
__global__ __launch_bounds__(32, 1) void query_mlp(
    const _Float16* __restrict__ feat16, const float* __restrict__ coord,
    const float* __restrict__ cell, const unsigned* __restrict__ w1qp,
    const unsigned* __restrict__ w2qp, const float* __restrict__ kk_b1,
    const float* __restrict__ kk_b2, const float* __restrict__ vv_b1,
    const float* __restrict__ vv_b2, float* __restrict__ out) {
  __shared__ unsigned hidu[2][2][16 * 128];  // [sample][mtile] 16x256 f16
  __shared__ int s_idx[2][32];
  __shared__ float s_r0[2][32], s_r1[2][32], s_w[2][32], s_rc0[32], s_rc1[32];

  const int lane = threadIdx.x;
  const int m = lane & 15, hb = lane >> 4;
  const int t = blockIdx.x;
  const int g = blockIdx.y;
  const float* b1 = g ? vv_b1 : kk_b1;
  const float* b2 = g ? vv_b2 : kk_b2;
  const unsigned* W1 = w1qp + g * (3 * 16 * 256);
  const unsigned* W2 = w2qp + g * (8 * 36 * 256);

  {  // geometry: each of the 32 lanes handles one query
    int j = t * 32 + lane;
    float c0 = coord[2 * j], c1 = coord[2 * j + 1];
    float ce0 = cell[2 * j], ce1 = cell[2 * j + 1];
    const float rr = 1.f / 128.f, eps = 1e-6f;
    float area[2];
#pragma unroll
    for (int s = 0; s < 2; ++s) {
      float vy = s ? 1.f : -1.f;
      float cy = fminf(fmaxf(c0 - rr + eps, -1.f + eps), 1.f - eps);
      float cx = fminf(fmaxf(c1 + vy * rr + eps, -1.f + eps), 1.f - eps);
      int iy = (int)fminf(fmaxf(roundf(((cy + 1.f) * 128.f - 1.f) * 0.5f), 0.f), 127.f);
      int ix = (int)fminf(fmaxf(roundf(((cx + 1.f) * 128.f - 1.f) * 0.5f), 0.f), 127.f);
      s_idx[s][lane] = iy * 128 + ix;
      float qcy = -1.f + (2.f * iy + 1.f) * rr;
      float qcx = -1.f + (2.f * ix + 1.f) * rr;
      float r0 = (c0 - qcy) * 128.f, r1 = (c1 - qcx) * 128.f;
      s_r0[s][lane] = r0;
      s_r1[s][lane] = r1;
      area[s] = fabsf(r0 * r1) + 1e-9f;
    }
    float tot = area[0] + area[1];
    s_w[0][lane] = area[0] / tot;
    s_w[1][lane] = area[1] / tot;
    s_rc0[lane] = ce0 * 128.f;
    s_rc1[lane] = ce1 * 128.f;
  }
  __syncthreads();

  // GEMM1 per sample: both M-tiles share each B fragment
  for (int s = 0; s < 2; ++s) {
    v16h A[2][3];  // [mtile][ktile]
#pragma unroll
    for (int mt = 0; mt < 2; ++mt) {
      const int q = mt * 16 + m;  // query row within the 32-group
      const unsigned* fr = (const unsigned*)feat16 + s_idx[s][q] * 32;
#pragma unroll
      for (int kt = 0; kt < 2; ++kt) A[mt][kt] = loadA_row(fr + kt * 16, hb);
      F16x16 G;  // ktile 2: geometry (K=64..67) + zero pad
#pragma unroll
      for (int i = 0; i < 8; ++i) {
        int K0 = hb * 16 + 2 * i;
#pragma unroll
        for (int ee = 0; ee < 2; ++ee) {
          int l = K0 + ee;
          float v = 0.f;
          if (l == 0) v = s_r0[s][q];
          else if (l == 1) v = s_r1[s][q];
          else if (l == 2) v = s_rc0[q];
          else if (l == 3) v = s_rc1[q];
          G.h[2 * i + ee] = (_Float16)v;
        }
      }
      A[mt][2] = G.v;
    }
    for (int nt = 0; nt < 16; ++nt) {
      v8f acc0 = (v8f){0.f, 0.f, 0.f, 0.f, 0.f, 0.f, 0.f, 0.f};
      v8f acc1 = (v8f){0.f, 0.f, 0.f, 0.f, 0.f, 0.f, 0.f, 0.f};
#pragma unroll
      for (int kt = 0; kt < 3; ++kt) {
        v16h B = loadB_swz(W1, 16, kt, nt, lane);
        acc0 = wmma_f16(A[0][kt], B, acc0);
        acc1 = wmma_f16(A[1][kt], B, acc1);
      }
      float bb = b1[nt * 16 + m];
      _Float16* h0 = (_Float16*)hidu[s][0];
      _Float16* h1 = (_Float16*)hidu[s][1];
#pragma unroll
      for (int r = 0; r < 8; ++r) {
        h0[(hb * 8 + r) * 256 + nt * 16 + m] = (_Float16)fmaxf(acc0[r] + bb, 0.f);
        h1[(hb * 8 + r) * 256 + nt * 16 + m] = (_Float16)fmaxf(acc1[r] + bb, 0.f);
      }
    }
  }
  __syncthreads();

  // per-row blend weights (C/D layout: row = hb*8 + r) per M-tile
  float w0r[2][8], w1r[2][8];
#pragma unroll
  for (int mt = 0; mt < 2; ++mt)
#pragma unroll
    for (int r = 0; r < 8; ++r) {
      w0r[mt][r] = s_w[0][mt * 16 + hb * 8 + r];
      w1r[mt][r] = s_w[1][mt * 16 + hb * 8 + r];
    }

  // preload GEMM2 A-fragments: [sample][mtile][ktile], reused over 36 N-tiles
  v16h A2[2][2][8];
#pragma unroll
  for (int s = 0; s < 2; ++s)
#pragma unroll
    for (int mt = 0; mt < 2; ++mt)
#pragma unroll
      for (int kt = 0; kt < 8; ++kt)
        A2[s][mt][kt] = loadA_row(hidu[s][mt] + m * 128 + kt * 16, hb);

  for (int nt = 0; nt < 36; ++nt) {
    if (nt + 1 < 36)  // warm next ntile's fragment column
      __builtin_prefetch(&W2[((nt + 1) * 32 + lane) * 8], 0, 1);
    v8f accA = (v8f){0.f, 0.f, 0.f, 0.f, 0.f, 0.f, 0.f, 0.f};  // s0,mt0
    v8f accB = (v8f){0.f, 0.f, 0.f, 0.f, 0.f, 0.f, 0.f, 0.f};  // s0,mt1
    v8f accC = (v8f){0.f, 0.f, 0.f, 0.f, 0.f, 0.f, 0.f, 0.f};  // s1,mt0
    v8f accD = (v8f){0.f, 0.f, 0.f, 0.f, 0.f, 0.f, 0.f, 0.f};  // s1,mt1
    v16h Bc = loadB_swz(W2, 36, 0, nt, lane);
#pragma unroll
    for (int kt = 0; kt < 8; ++kt) {
      v16h Bn = Bc;
      if (kt < 7) Bn = loadB_swz(W2, 36, kt + 1, nt, lane);  // issue ahead
      accA = wmma_f16(A2[0][0][kt], Bc, accA);
      accB = wmma_f16(A2[0][1][kt], Bc, accB);
      accC = wmma_f16(A2[1][0][kt], Bc, accC);
      accD = wmma_f16(A2[1][1][kt], Bc, accD);
      Bc = Bn;
    }
    float bb = b2[nt * 16 + m];
#pragma unroll
    for (int r = 0; r < 8; ++r) {
      float v0 = w0r[0][r] * accA[r] + w1r[0][r] * accC[r] + bb;
      float v1 = w0r[1][r] * accB[r] + w1r[1][r] * accD[r] + bb;
      int row0 = t * 32 + hb * 8 + r;
      int row1 = t * 32 + 16 + hb * 8 + r;
      out[row0 * 1152 + g * 576 + nt * 16 + m] = v0;
      out[row1 * 1152 + g * 576 + nt * 16 + m] = v1;
    }
  }
}

// ---------------------------------------------------------------------------
extern "C" void kernel_launch(void* const* d_in, const int* in_sizes, int n_in,
                              void* d_out, int out_size, void* d_ws,
                              size_t ws_size, hipStream_t stream) {
  const float* feat_i = (const float*)d_in[0];
  const float* coord = (const float*)d_in[1];
  const float* cell = (const float*)d_in[2];
  const float* coef_w = (const float*)d_in[3];
  const float* coef_b = (const float*)d_in[4];
  const float* freq_w = (const float*)d_in[5];
  const float* freq_b = (const float*)d_in[6];
  const float* phase_w = (const float*)d_in[7];
  const float* phase_b = (const float*)d_in[8];
  const float* im_w1 = (const float*)d_in[9];
  const float* im_b1 = (const float*)d_in[10];
  const float* im_w2 = (const float*)d_in[11];
  const float* im_b2 = (const float*)d_in[12];
  const float* kk_w1 = (const float*)d_in[13];
  const float* kk_b1 = (const float*)d_in[14];
  const float* kk_w2 = (const float*)d_in[15];
  const float* kk_b2 = (const float*)d_in[16];
  const float* vv_w1 = (const float*)d_in[17];
  const float* vv_b1 = (const float*)d_in[18];
  const float* vv_w2 = (const float*)d_in[19];
  const float* vv_b2 = (const float*)d_in[20];

  char* ws = (char*)d_ws;
  unsigned* cwp = (unsigned*)(ws + 0);            // 18*10*256*4 = 184320
  unsigned* w1p = (unsigned*)(ws + 184320);       // 2*16*256*4  = 32768
  unsigned* w2p = (unsigned*)(ws + 217088);       // 8*4*256*4   = 32768
  unsigned* w1qp = (unsigned*)(ws + 249856);      // 2*3*16*256*4= 98304
  unsigned* w2qp = (unsigned*)(ws + 348160);      // 2*8*36*256*4= 589824
  _Float16* x16 = (_Float16*)(ws + 937984);       // 16384*64*2  = 2097152
  _Float16* feat16 = (_Float16*)(ws + 3035136);   // 16384*64*2  = 2097152
  float* out = (float*)d_out;

  const int prep_elems = 18 * 10 * 256 + 2 * 16 * 256 + 8 * 4 * 256 +
                         2 * 3 * 16 * 256 + 2 * 8 * 36 * 256;
  prep_weights<<<(prep_elems + 255) / 256, 256, 0, stream>>>(
      coef_w, freq_w, phase_w, im_w1, im_w2, kk_w1, kk_w2, vv_w1, vv_w2,
      cwp, w1p, w2p, w1qp, w2qp);
  conv_lte<<<1024, 32, 0, stream>>>(feat_i, cwp, coef_b, freq_b, phase_b, x16);
  imnet_mlp<<<1024, 32, 0, stream>>>(x16, w1p, w2p, im_b1, im_b2, feat16);
  query_mlp<<<dim3(2048, 2), 32, 0, stream>>>(feat16, coord, cell, w1qp, w2qp,
                                              kk_b1, kk_b2, vv_b1, vv_b2, out);
}
